// Gtransformerblock_58085137711797
// MI455X (gfx1250) — compile-verified, matrix operand order checked by reading, not run on hardware
//
#include <hip/hip_runtime.h>

// ---------------------------------------------------------------------------
// Graph-transformer block on MI455X (gfx1250), wave32 + WMMA bf16 16x16x32.
// Dominant cost: one streaming pass over adj (256 MB, NT loads). All matmuls
// run on v_wmma_f32_16x16x32_bf16. Softmax uses the exact identity
//   softmax(masked scores) -> (sum_edges (e^s - 1) v + sum_all v) /
//                             (sum_edges (e^s - 1) + N)
// which removes max-tracking and makes accumulation fully associative.
// ---------------------------------------------------------------------------

typedef __attribute__((ext_vector_type(16))) __bf16 v16bf;
typedef __attribute__((ext_vector_type(8)))  __bf16 v8bf;
typedef __attribute__((ext_vector_type(8)))  float  v8f;
typedef __attribute__((ext_vector_type(4)))  float  v4f;

#define DEVINL __device__ __forceinline__

DEVINL v8f wmma_bf16(v16bf a, v16bf b, v8f c) {
  // D = A(16x32) * B(32x16) + C(16x16), f32 accum
  return __builtin_amdgcn_wmma_f32_16x16x32_bf16(false, a, false, b, (short)0, c,
                                                 false, false);
}

// Fragment element mapping (A and B use the same per-lane K mapping):
//   lane L: row/col = L&15 ; element e -> k = (L>>4)*8 + e + (e>=8 ? 8 : 0)
// For a row-major source this is two contiguous 16B chunks at +0 and +32B.
DEVINL v16bf load_frag_bf16(const __bf16* rowk, int lh) {
  const __bf16* p = rowk + lh * 8;
  v8bf lo = *(const v8bf*)(p);
  v8bf hi = *(const v8bf*)(p + 16);
  v16bf r;
#pragma unroll
  for (int i = 0; i < 8; ++i) { r[i] = lo[i]; r[i + 8] = hi[i]; }
  return r;
}

DEVINL v16bf load_frag_f32(const float* rowk, int lh) {
  const float* p = rowk + lh * 8;
  v4f a0 = *(const v4f*)(p);
  v4f a1 = *(const v4f*)(p + 4);
  v4f b0 = *(const v4f*)(p + 16);
  v4f b1 = *(const v4f*)(p + 20);
  v16bf r;
#pragma unroll
  for (int i = 0; i < 4; ++i) {
    r[i]      = (__bf16)a0[i];
    r[i + 4]  = (__bf16)a1[i];
    r[i + 8]  = (__bf16)b0[i];
    r[i + 12] = (__bf16)b1[i];
  }
  return r;
}

// ---------------------------------------------------------------------------
// Prep: convert + transpose weights to bf16.
//   Wqt/Wkt/Wvt[h][f][d] (f=64,d=256), W1t[512][256], W2t[256][512]
// ---------------------------------------------------------------------------
__global__ void k_prep(const float* __restrict__ Wq, const float* __restrict__ Wk,
                       const float* __restrict__ Wv, const float* __restrict__ W1,
                       const float* __restrict__ W2, __bf16* Wqt, __bf16* Wkt,
                       __bf16* Wvt, __bf16* W1t, __bf16* W2t) {
  int i = blockIdx.x * 256 + threadIdx.x;
  if (i < 196608) {                 // 3 * 4*64*256
    int sel = i / 65536;
    int r = i % 65536;
    int h = r >> 14;
    int f = (r >> 8) & 63;
    int d = r & 255;
    const float* src = (sel == 0) ? Wq : (sel == 1) ? Wk : Wv;
    __bf16* dst = (sel == 0) ? Wqt : (sel == 1) ? Wkt : Wvt;
    dst[r] = (__bf16)src[(h * 256 + d) * 64 + f];
  } else if (i < 327680) {          // W1t [512][256]
    int r = i - 196608;
    int o = r >> 8, kk = r & 255;
    W1t[r] = (__bf16)W1[kk * 512 + o];
  } else {                          // W2t [256][512]
    int r = i - 327680;
    int o = r >> 9, kk = r & 511;
    W2t[r] = (__bf16)W2[kk * 256 + o];
  }
}

// ---------------------------------------------------------------------------
// QKV projections. One wave per 16x16 output tile.
//   q[h][n][64], k[h][m][64] (row-major bf16), vT[h][64][4096] (transposed).
//   grid id = (((sel*4 + h)*256 + nb)*4 + dt)
// ---------------------------------------------------------------------------
__global__ void k_qkv(const float* __restrict__ hin, const __bf16* __restrict__ Wqt,
                      const __bf16* __restrict__ Wkt, const __bf16* __restrict__ Wvt,
                      __bf16* qb, __bf16* kb, __bf16* vT) {
  int id = blockIdx.x;
  int dt = id & 3;  id >>= 2;
  int nb = id & 255; id >>= 8;
  int h  = id & 3;  id >>= 2;
  int sel = id;
  int lane = threadIdx.x, lh = lane >> 4, ln = lane & 15;

  const float* arow = hin + (size_t)(nb * 16 + ln) * 256;
  const __bf16* Wt = (sel == 0) ? Wqt : (sel == 1) ? Wkt : Wvt;
  const __bf16* brow = Wt + (size_t)(h * 64 + dt * 16 + ln) * 256;

  v8f c = {};
#pragma unroll
  for (int kk = 0; kk < 8; ++kk)
    c = wmma_bf16(load_frag_f32(arow + kk * 32, lh),
                  load_frag_bf16(brow + kk * 32, lh), c);

  if (sel < 2) {  // row-major [h][n][64]; C: lane<->col f, elem<->row n
    __bf16* out = ((sel == 0) ? qb : kb) + (size_t)(h * 4096 + nb * 16) * 64 + dt * 16 + ln;
#pragma unroll
    for (int j = 0; j < 8; ++j) out[(size_t)(j + 8 * lh) * 64] = (__bf16)c[j];
  } else {        // vT[h][f][m]: per-lane 8 consecutive m -> one 16B store
    v8bf pk;
#pragma unroll
    for (int j = 0; j < 8; ++j) pk[j] = (__bf16)c[j];
    *(v8bf*)(vT + (size_t)(h * 64 + dt * 16 + ln) * 4096 + nb * 16 + 8 * lh) = pk;
  }
}

// ---------------------------------------------------------------------------
// Vtot[h*64+f] = sum_m v[m][f]  (reduce one vT row per wave)
// ---------------------------------------------------------------------------
__global__ void k_vtot(const __bf16* __restrict__ vT, float* Vtot) {
  int hf = blockIdx.x;
  int lane = threadIdx.x;
  const __bf16* row = vT + (size_t)hf * 4096;
  float s = 0.f;
  for (int m = lane; m < 4096; m += 32) s += (float)row[m];
#pragma unroll
  for (int off = 16; off; off >>= 1) s += __shfl_xor(s, off, 32);
  if (lane == 0) Vtot[hf] = s;
}

// ---------------------------------------------------------------------------
// Attention, single streaming pass over adj.
// Per (h, 16-row block ib, m-partition part): accumulate
//   xT[f,n] += sum_m vT[f,m] * w[n,m],  den[n] += sum_m w[n,m]
// where w = exp(qk*scale*adj) - 1. S^T = k·q^T is computed so its C fragment
// is directly reusable (after f32->bf16 cvt) as the B fragment of w^T.
//   grid id = (h*256 + ib)*4 + part
// ---------------------------------------------------------------------------
__global__ void k_attn(const __bf16* __restrict__ qb, const __bf16* __restrict__ kb,
                       const __bf16* __restrict__ vT, const float* __restrict__ adj,
                       float* xpart, float* denpart) {
  int bid = blockIdx.x;
  int part = bid & 3;
  int hib = bid >> 2;
  int h = hib >> 8, ib = hib & 255;
  int lane = threadIdx.x, lh = lane >> 4, ln = lane & 15;
  const float scale = 0.0625f;  // 1/sqrt(256)

  // B_q fragments (q^T, fixed over the m loop): lane<->n, elems<->d
  const __bf16* qrow = qb + (size_t)(h * 4096 + ib * 16 + ln) * 64;
  v16bf Bq0 = load_frag_bf16(qrow, lh);
  v16bf Bq1 = load_frag_bf16(qrow + 32, lh);

  const __bf16* kh  = kb + (size_t)h * 4096 * 64;
  const __bf16* vTh = vT + (size_t)h * 64 * 4096;
  // adj row for this lane's n, offset to its m half
  const float* adjrow = adj + ((size_t)h * 4096 + ib * 16 + ln) * 4096 + lh * 8;

  v8f Cx0 = {}, Cx1 = {}, Cx2 = {}, Cx3 = {};
  float den = 0.f;

  for (int mt = part * 32; mt < part * 32 + 32; ++mt) {
    int m0 = mt * 32;
    // Two 16x16 S^T sub-tiles (K = d = 64 -> 2 WMMAs each)
    const __bf16* kr0 = kh + (size_t)(m0 + ln) * 64;
    const __bf16* kr1 = kh + (size_t)(m0 + 16 + ln) * 64;
    v8f s0 = {}, s1 = {};
    s0 = wmma_bf16(load_frag_bf16(kr0,      lh), Bq0, s0);
    s0 = wmma_bf16(load_frag_bf16(kr0 + 32, lh), Bq1, s0);
    s1 = wmma_bf16(load_frag_bf16(kr1,      lh), Bq0, s1);
    s1 = wmma_bf16(load_frag_bf16(kr1 + 32, lh), Bq1, s1);

    // adj: per-lane contiguous floats (streamed once -> non-temporal)
    const float* ap = adjrow + m0;
    v4f a0 = __builtin_nontemporal_load((const v4f*)(ap));
    v4f a1 = __builtin_nontemporal_load((const v4f*)(ap + 4));
    v4f a2 = __builtin_nontemporal_load((const v4f*)(ap + 16));
    v4f a3 = __builtin_nontemporal_load((const v4f*)(ap + 20));

    // w = exp(s*scale*adj) - 1 ; pack straight into a B fragment of w^T
    v16bf bw;
#pragma unroll
    for (int j = 0; j < 4; ++j) {
      float w0 = __expf(s0[j]     * scale * a0[j]) - 1.0f;
      float w1 = __expf(s0[j + 4] * scale * a1[j]) - 1.0f;
      float w2 = __expf(s1[j]     * scale * a2[j]) - 1.0f;
      float w3 = __expf(s1[j + 4] * scale * a3[j]) - 1.0f;
      den += (w0 + w1) + (w2 + w3);
      bw[j]      = (__bf16)w0;
      bw[j + 4]  = (__bf16)w1;
      bw[j + 8]  = (__bf16)w2;
      bw[j + 12] = (__bf16)w3;
    }

    // xT[f-tile] += vT_tile(16f x 32m) * w^T(32m x 16n)
    const __bf16* vr = vTh + (size_t)ln * 4096 + m0;
    Cx0 = wmma_bf16(load_frag_bf16(vr,             lh), bw, Cx0);
    Cx1 = wmma_bf16(load_frag_bf16(vr + 16 * 4096, lh), bw, Cx1);
    Cx2 = wmma_bf16(load_frag_bf16(vr + 32 * 4096, lh), bw, Cx2);
    Cx3 = wmma_bf16(load_frag_bf16(vr + 48 * 4096, lh), bw, Cx3);
  }

  // raw per-lane dump of partial numerator + denominator (deterministic combine)
  float* xp = xpart + ((size_t)bid * 32 + lane) * 32;
  *(v8f*)(xp)      = Cx0;
  *(v8f*)(xp + 8)  = Cx1;
  *(v8f*)(xp + 16) = Cx2;
  *(v8f*)(xp + 24) = Cx3;
  denpart[(size_t)bid * 32 + lane] = den;
}

// ---------------------------------------------------------------------------
// Combine partitions, add sum_m v, normalize, emit emb (bf16 [4096][256]).
// ---------------------------------------------------------------------------
__global__ void k_combine(const float* __restrict__ xpart,
                          const float* __restrict__ denpart,
                          const float* __restrict__ Vtot, __bf16* emb) {
  int hib = blockIdx.x;
  int h = hib >> 8, ib = hib & 255;
  int lane = threadIdx.x, lh = lane >> 4, ln = lane & 15;
  v8f A0 = {}, A1 = {}, A2 = {}, A3 = {};
  float den = 0.f;
#pragma unroll
  for (int p = 0; p < 4; ++p) {
    const float* xp = xpart + ((size_t)(hib * 4 + p) * 32 + lane) * 32;
    A0 += *(const v8f*)(xp);
    A1 += *(const v8f*)(xp + 8);
    A2 += *(const v8f*)(xp + 16);
    A3 += *(const v8f*)(xp + 24);
    den += denpart[(size_t)(hib * 4 + p) * 32 + lane];
  }
  den += __shfl_xor(den, 16, 32);            // lane halves hold complementary m
  float inv = 1.0f / (den + 4096.0f);        // + N non-edge exp(0) terms
  __bf16* erow = emb + (size_t)(ib * 16 + ln) * 256 + h * 64 + lh * 8;
  v8f acc[4] = {A0, A1, A2, A3};
#pragma unroll
  for (int ft = 0; ft < 4; ++ft) {
    v8bf pk;
#pragma unroll
    for (int j = 0; j < 8; ++j) {
      int f = ft * 16 + lh * 8 + j;
      pk[j] = (__bf16)((acc[ft][j] + Vtot[h * 64 + f]) * inv);
    }
    *(v8bf*)(erow + ft * 16) = pk;
  }
}

// ---------------------------------------------------------------------------
// MLP1: p1 = relu(emb @ W1 + b1) -> bf16 [4096][512]. One wave per 16x16 tile.
// ---------------------------------------------------------------------------
__global__ void k_mlp1(const __bf16* __restrict__ emb, const __bf16* __restrict__ W1t,
                       const float* __restrict__ b1, __bf16* p1) {
  int id = blockIdx.x;
  int ct = id & 31, nb = id >> 5;
  int lane = threadIdx.x, lh = lane >> 4, ln = lane & 15;
  const __bf16* arow = emb + (size_t)(nb * 16 + ln) * 256;
  const __bf16* brow = W1t + (size_t)(ct * 16 + ln) * 256;
  v8f c = {};
#pragma unroll
  for (int kk = 0; kk < 8; ++kk)
    c = wmma_bf16(load_frag_bf16(arow + kk * 32, lh),
                  load_frag_bf16(brow + kk * 32, lh), c);
  float bias = b1[ct * 16 + ln];
  __bf16* out = p1 + (size_t)(nb * 16) * 512 + ct * 16 + ln;
#pragma unroll
  for (int j = 0; j < 8; ++j)
    out[(size_t)(j + 8 * lh) * 512] = (__bf16)fmaxf(c[j] + bias, 0.0f);
}

// ---------------------------------------------------------------------------
// MLP2 + row softmax: 4 waves per 16-row block; p2 tile staged in LDS.
// ---------------------------------------------------------------------------
__global__ void k_mlp2(const __bf16* __restrict__ p1, const __bf16* __restrict__ W2t,
                       const float* __restrict__ b2, float* __restrict__ out) {
  __shared__ float sm[16 * 260];  // pad stride to dodge bank conflicts
  int nb = blockIdx.x;
  int tid = threadIdx.x;
  int wave = tid >> 5, lane = tid & 31, lh = lane >> 4, ln = lane & 15;
  const __bf16* arow = p1 + (size_t)(nb * 16 + ln) * 512;
#pragma unroll
  for (int ct = 0; ct < 4; ++ct) {
    int col = wave * 64 + ct * 16 + ln;
    const __bf16* brow = W2t + (size_t)col * 512;
    v8f c = {};
#pragma unroll
    for (int kk = 0; kk < 16; ++kk)
      c = wmma_bf16(load_frag_bf16(arow + kk * 32, lh),
                    load_frag_bf16(brow + kk * 32, lh), c);
    float bias = b2[col];
#pragma unroll
    for (int j = 0; j < 8; ++j) sm[(j + 8 * lh) * 260 + col] = c[j] + bias;
  }
  __syncthreads();
  if (tid < 16) {
    int r = tid;
    float mx = -3.0e38f;
    for (int cc = 0; cc < 256; ++cc) mx = fmaxf(mx, sm[r * 260 + cc]);
    float s = 0.f;
    for (int cc = 0; cc < 256; ++cc) {
      float e = __expf(sm[r * 260 + cc] - mx);
      sm[r * 260 + cc] = e;
      s += e;
    }
    float inv = 1.0f / s;
    float* orow = out + (size_t)(nb * 16 + r) * 256;
    for (int cc = 0; cc < 256; ++cc) orow[cc] = sm[r * 260 + cc] * inv;
  }
}

// ---------------------------------------------------------------------------
extern "C" void kernel_launch(void* const* d_in, const int* in_sizes, int n_in,
                              void* d_out, int out_size, void* d_ws, size_t ws_size,
                              hipStream_t stream) {
  (void)in_sizes; (void)n_in; (void)out_size; (void)ws_size;
  const float* hin = (const float*)d_in[0];
  const float* adj = (const float*)d_in[1];
  const float* Wq  = (const float*)d_in[2];
  const float* Wk  = (const float*)d_in[3];
  const float* Wv  = (const float*)d_in[4];
  const float* W1  = (const float*)d_in[5];
  const float* b1  = (const float*)d_in[6];
  const float* W2  = (const float*)d_in[7];
  const float* b2  = (const float*)d_in[8];
  float* out = (float*)d_out;

  char* ws = (char*)d_ws;
  size_t off = 0;
  auto alloc = [&](size_t bytes) -> void* {
    void* p = ws + off;
    off = (off + bytes + 255) & ~(size_t)255;
    return p;
  };
  __bf16* Wqt  = (__bf16*)alloc(65536 * 2);
  __bf16* Wkt  = (__bf16*)alloc(65536 * 2);
  __bf16* Wvt  = (__bf16*)alloc(65536 * 2);
  __bf16* W1t  = (__bf16*)alloc(131072 * 2);
  __bf16* W2t  = (__bf16*)alloc(131072 * 2);
  __bf16* qb   = (__bf16*)alloc((size_t)4 * 4096 * 64 * 2);
  __bf16* kb   = (__bf16*)alloc((size_t)4 * 4096 * 64 * 2);
  __bf16* vT   = (__bf16*)alloc((size_t)4 * 64 * 4096 * 2);
  float*  Vtot = (float*)alloc(256 * 4);
  __bf16* emb  = (__bf16*)alloc((size_t)4096 * 256 * 2);
  __bf16* p1   = (__bf16*)alloc((size_t)4096 * 512 * 2);
  float*  xpart   = (float*)alloc((size_t)4096 * 32 * 32 * 4);  // 16 MB
  float*  denpart = (float*)alloc((size_t)4096 * 32 * 4);

  k_prep<<<1792, 256, 0, stream>>>(Wq, Wk, Wv, W1, W2, Wqt, Wkt, Wvt, W1t, W2t);
  k_qkv<<<12288, 32, 0, stream>>>(hin, Wqt, Wkt, Wvt, qb, kb, vT);
  k_vtot<<<256, 32, 0, stream>>>(vT, Vtot);
  k_attn<<<4096, 32, 0, stream>>>(qb, kb, vT, adj, xpart, denpart);
  k_combine<<<1024, 32, 0, stream>>>(xpart, denpart, Vtot, emb);
  k_mlp1<<<8192, 32, 0, stream>>>(emb, W1t, b1, p1);
  k_mlp2<<<256, 128, 0, stream>>>(p1, W2t, b2, out);
}
